// MedicalEncoder_77094662963356
// MI455X (gfx1250) — compile-verified, matrix operand order checked by reading, not run on hardware
//
#include <hip/hip_runtime.h>

// ---------------------------------------------------------------------------
// MedicalEncoder for MI455X (gfx1250, wave32).
// All GEMMs: V_WMMA_F32_16X16X32_BF16, f32 accumulate. Operands are
// pre-converted to bf16 (weights also pre-transposed to N x K) so every
// fragment is two contiguous b128 loads. One wave computes a 16x64 C tile
// (4 accumulators, A fragment reused 4x), with a software-pipelined K-loop
// (next K-step's loads issued before current WMMAs). Edge gathers are fused
// into the A-operand row indexing and served by the 192MB L2 (node feature
// buffers are 23MB each). Residual rows are prefetched (global_prefetch_b8).
// ---------------------------------------------------------------------------

typedef __attribute__((ext_vector_type(16))) __bf16 v16bf;
typedef __attribute__((ext_vector_type(8)))  float  v8f;

#define NMSG 10000
#define NCON 20000
#define NNODE 30000
#define DM 384
#define NH 6
#define DH 64
#define NE 128000

// ----------------------------- utility kernels -----------------------------

__global__ void fill_kernel(float* __restrict__ p, float v, long long n) {
  long long i = (long long)blockIdx.x * blockDim.x + threadIdx.x;
  long long st = (long long)gridDim.x * blockDim.x;
  for (; i < n; i += st) p[i] = v;
}

// y_bf16[i] = gelu_exact(x[i])
__global__ void gelu_bf16_kernel(const float* __restrict__ x,
                                 __bf16* __restrict__ y, long long n) {
  long long i = (long long)blockIdx.x * blockDim.x + threadIdx.x;
  long long st = (long long)gridDim.x * blockDim.x;
  for (; i < n; i += st) {
    float v = x[i];
    y[i] = (__bf16)(0.5f * v * (1.0f + erff(v * 0.70710678118654752f)));
  }
}

// out[b][n*K+k] = (bf16) in[b][k*N+n]   (weight convert + transpose)
__global__ void transpose_cvt_kernel(const float* __restrict__ in,
                                     __bf16* __restrict__ out, int K, int N) {
  const long long tot = (long long)K * N;
  const float* ib = in + (size_t)blockIdx.y * tot;
  __bf16* ob = out + (size_t)blockIdx.y * tot;
  long long i = (long long)blockIdx.x * blockDim.x + threadIdx.x;
  const long long st = (long long)gridDim.x * blockDim.x;
  for (; i < tot; i += st) {
    const int n = (int)(i / K);
    const int k = (int)(i - (long long)n * K);
    ob[i] = (__bf16)ib[(size_t)k * N + n];
  }
}

// masked mean pool over L=16 tokens then L2 normalize; bf16 output.
__global__ void pool_norm_kernel(const float* __restrict__ tok,
                                 const int* __restrict__ ids,
                                 __bf16* __restrict__ out) {
  const int n = blockIdx.x;
  __shared__ float red[128];
  __shared__ float s_inv;
  const int* idn = ids + (size_t)n * 16;
  float mask[16];
  float cnt = 0.f;
#pragma unroll
  for (int l = 0; l < 16; ++l) { mask[l] = (idn[l] > 0) ? 1.f : 0.f; cnt += mask[l]; }
  const float inv_cnt = 1.0f / fmaxf(cnt, 1e-9f);
  const float* t = tok + (size_t)n * 16 * DM;
  float vals[3];
  float ss = 0.f;
#pragma unroll
  for (int j = 0; j < 3; ++j) {
    const int d = threadIdx.x + j * 128;
    float s = 0.f;
#pragma unroll
    for (int l = 0; l < 16; ++l) s += t[(size_t)l * DM + d] * mask[l];
    s *= inv_cnt;
    vals[j] = s;
    ss += s * s;
  }
  red[threadIdx.x] = ss;
  __syncthreads();
  for (int off = 64; off > 0; off >>= 1) {
    if (threadIdx.x < off) red[threadIdx.x] += red[threadIdx.x + off];
    __syncthreads();
  }
  if (threadIdx.x == 0) s_inv = 1.0f / fmaxf(sqrtf(red[0]), 1e-12f);
  __syncthreads();
#pragma unroll
  for (int j = 0; j < 3; ++j)
    out[(size_t)n * DM + threadIdx.x + j * 128] = (__bf16)(vals[j] * s_inv);
}

// ------------------------- unified bf16 WMMA GEMM ---------------------------
// CDNA5 16-bit A fragment: lane m=lane&15, elem i holds K=i+8*(i>=8)+8*(lane>=16)
// -> per lane: two contiguous 8xbf16 (16B) runs at K-offsets {koff, koff+16}.
struct Frag32B { uint4 lo, hi; };
__device__ inline v16bf load_frag(const __bf16* p) {
  Frag32B t;
  t.lo = *reinterpret_cast<const uint4*>(p);
  t.hi = *reinterpret_cast<const uint4*>(p + 16);
  return __builtin_bit_cast(v16bf, t);
}

// Y = epilogue( A[rowIdx? rowBase+rowIdx[m] : m, :] @ Wt^T + bias )
// Wt is pre-transposed (N x K, bf16). One wave: 16x64 tile, 4 accumulators.
// act: 0=none, 1=relu, 2=skip-blend a=sigmoid(*skipPtr): y=a*(xw+b)+(1-a)*resid
// Output: f32 to Yf (if non-null) and/or bf16 to Yb (if non-null).
// Requires K a multiple of 32 and K >= 32 (here K is 64 or 384).
__global__ void gemm_bf16_wmma(const __bf16* __restrict__ A, int lda,
                               const int* __restrict__ rowIdx, int rowBase,
                               const __bf16* __restrict__ Wt, int ldw,
                               const float* __restrict__ bias,
                               const float* __restrict__ resid, int ldres,
                               const float* __restrict__ skipPtr,
                               float* __restrict__ Yf, __bf16* __restrict__ Yb,
                               int ldy, int M, int N, int K, int act,
                               int aHeadOff, int wHeadOff, int yHeadOff) {
  const int head = blockIdx.y;
  const __bf16* Ah = A + (size_t)head * aHeadOff;
  const __bf16* Wh = Wt + (size_t)head * wHeadOff;
  float*  Yfh = Yf ? (Yf + (size_t)head * yHeadOff) : nullptr;
  __bf16* Ybh = Yb ? (Yb + (size_t)head * yHeadOff) : nullptr;

  const int lane = threadIdx.x & 31;
  const int wave = threadIdx.x >> 5;
  const int ntilesN = N >> 6;                 // 64-wide wave tiles
  const int ntiles = ((M + 15) >> 4) * ntilesN;
  const int tile = blockIdx.x * (blockDim.x >> 5) + wave;
  if (tile >= ntiles) return;
  const int tm = tile / ntilesN;
  const int tn = tile - tm * ntilesN;
  const int row0 = tm << 4, col0 = tn << 6;

  const int m  = lane & 15;
  const int hi = lane >> 4;
  const int koff = hi << 3;                   // +8 K offset for lanes 16..31

  // prefetch the f32 residual tile (2 cache lines/row x 16 rows == 32 lanes)
  if (resid) {
    const int pr = row0 + m;
    const int pc = col0 + hi * 32;
    if (pr < M) __builtin_prefetch(&resid[(size_t)pr * ldres + pc], 0, 3);
  }

  int arow = row0 + m;
  if (arow >= M) arow = M - 1;                // clamp (stores are guarded)
  const int xrow = rowIdx ? (rowBase + rowIdx[arow]) : arow;
  const __bf16* aptr = Ah + (size_t)xrow * lda + koff;
  const __bf16* wptr = Wh + (size_t)(col0 + m) * ldw + koff;

  v8f acc[4] = {{}, {}, {}, {}};
  // software pipeline: preload fragments for K-step 0
  v16bf af = load_frag(aptr);
  v16bf bfv[4];
#pragma unroll
  for (int j = 0; j < 4; ++j) bfv[j] = load_frag(wptr + (size_t)(16 * j) * ldw);
  for (int k0 = 32; k0 < K; k0 += 32) {
    // issue next K-step's loads before consuming current fragments
    const v16bf naf = load_frag(aptr + k0);
    v16bf nbf[4];
#pragma unroll
    for (int j = 0; j < 4; ++j)
      nbf[j] = load_frag(wptr + (size_t)(16 * j) * ldw + k0);
#pragma unroll
    for (int j = 0; j < 4; ++j)
      acc[j] = __builtin_amdgcn_wmma_f32_16x16x32_bf16(
          false, af, false, bfv[j], (short)0, acc[j], false, false);
    af = naf;
#pragma unroll
    for (int j = 0; j < 4; ++j) bfv[j] = nbf[j];
  }
#pragma unroll
  for (int j = 0; j < 4; ++j)
    acc[j] = __builtin_amdgcn_wmma_f32_16x16x32_bf16(
        false, af, false, bfv[j], (short)0, acc[j], false, false);

  float aS = 1.f, bS = 0.f;
  if (skipPtr) { aS = 1.f / (1.f + __expf(-skipPtr[0])); bS = 1.f - aS; }
  const int rbase = row0 + (hi << 3);         // C/D: vgpr r -> row r+8*(lane>=16)
#pragma unroll
  for (int j = 0; j < 4; ++j) {
    const int col = col0 + 16 * j + m;
    const float bv = bias ? bias[col] : 0.f;
#pragma unroll
    for (int r = 0; r < 8; ++r) {
      const int rr = rbase + r;
      if (rr < M) {
        float v = acc[j][r] + bv;
        if (act == 1) v = fmaxf(v, 0.f);
        else if (act == 2) v = aS * v + bS * resid[(size_t)rr * ldres + col];
        if (Yfh) Yfh[(size_t)rr * ldy + col] = v;
        if (Ybh) Ybh[(size_t)rr * ldy + col] = (__bf16)v;
      }
    }
  }
}

// ------------------------- edge softmax / aggregate -------------------------

__device__ inline void atomicMaxFloat(float* addr, float val) {
  if (val >= 0.f) atomicMax(reinterpret_cast<int*>(addr), __float_as_int(val));
  else atomicMin(reinterpret_cast<unsigned int*>(addr), __float_as_uint(val));
}

__global__ void score_max_kernel(const float* __restrict__ qbuf, int dstBase,
                                 const int* __restrict__ dst,
                                 const float* __restrict__ ke,
                                 const float* __restrict__ prel,
                                 float* __restrict__ score,
                                 float* __restrict__ maxbuf, int E) {
  const int idx = blockIdx.x * blockDim.x + threadIdx.x;
  if (idx >= E * NH) return;
  const int e = idx / NH, h = idx - e * NH;
  const int d = dstBase + dst[e];
  const float4* qr = (const float4*)(qbuf + (size_t)d * DM + h * DH);
  const float4* kr = (const float4*)(ke + (size_t)e * DM + h * DH);
  float s = 0.f;
#pragma unroll
  for (int i = 0; i < 16; ++i) {
    float4 a = qr[i], b = kr[i];
    s += a.x * b.x + a.y * b.y + a.z * b.z + a.w * b.w;
  }
  s *= prel[h] * 0.125f;  // 1/sqrt(64)
  score[idx] = s;
  atomicMaxFloat(&maxbuf[d * NH + h], s);
}

__global__ void exp_den_kernel(const int* __restrict__ dst, int dstBase,
                               float* __restrict__ score,
                               const float* __restrict__ maxbuf,
                               float* __restrict__ den, int E) {
  const int idx = blockIdx.x * blockDim.x + threadIdx.x;
  if (idx >= E * NH) return;
  const int e = idx / NH, h = idx - e * NH;
  const int d = dstBase + dst[e];
  const float ex = __expf(score[idx] - maxbuf[d * NH + h]);
  score[idx] = ex;
  atomicAdd(&den[d * NH + h], ex);
}

__global__ void aggregate_kernel(const int* __restrict__ dst, int dstBase,
                                 const float* __restrict__ msgT,
                                 const float* __restrict__ score,
                                 const float* __restrict__ den,
                                 float* __restrict__ agg, long long total) {
  long long i = (long long)blockIdx.x * blockDim.x + threadIdx.x;
  long long st = (long long)gridDim.x * blockDim.x;
  for (; i < total; i += st) {
    const int e = (int)(i / DM);
    const int c = (int)(i - (long long)e * DM);
    const int h = c >> 6;
    const int d = dstBase + dst[e];
    const float alpha = score[e * NH + h] / den[d * NH + h];
    atomicAdd(&agg[(size_t)d * DM + c], msgT[i] * alpha);
  }
}

// ------------------------------- launcher ----------------------------------

static void launch_gemm(hipStream_t stream, const __bf16* A, int lda,
                        const int* rowIdx, int rowBase, const __bf16* Wt,
                        int ldw, const float* bias, const float* resid,
                        int ldres, const float* skipPtr, float* Yf, __bf16* Yb,
                        int ldy, int M, int N, int K, int act, int heads,
                        int aHeadOff, int wHeadOff, int yHeadOff) {
  const int tiles = ((M + 15) / 16) * (N / 64);
  dim3 grid((tiles + 7) / 8, heads, 1);
  gemm_bf16_wmma<<<grid, 256, 0, stream>>>(A, lda, rowIdx, rowBase, Wt, ldw,
                                           bias, resid, ldres, skipPtr, Yf, Yb,
                                           ldy, M, N, K, act, aHeadOff,
                                           wHeadOff, yHeadOff);
}

static void launch_fill(hipStream_t stream, float* p, float v, long long n) {
  int blocks = (int)((n + 255) / 256);
  if (blocks > 65535) blocks = 65535;
  fill_kernel<<<blocks, 256, 0, stream>>>(p, v, n);
}

extern "C" void kernel_launch(void* const* d_in, const int* in_sizes, int n_in,
                              void* d_out, int out_size, void* d_ws,
                              size_t ws_size, hipStream_t stream) {
  const float* tok_msg = (const float*)d_in[0];
  const float* tok_con = (const float*)d_in[1];
  const int*   ids_msg = (const int*)d_in[2];
  const int*   ids_con = (const int*)d_in[3];
  const int*   src_cm  = (const int*)d_in[4];
  const int*   dst_cm  = (const int*)d_in[5];
  const int*   src_mc  = (const int*)d_in[6];
  const int*   dst_mc  = (const int*)d_in[7];
  const float* lin_W   = (const float*)d_in[8];
  const float* lin_b   = (const float*)d_in[9];
  const float* kW      = (const float*)d_in[10];
  const float* kb      = (const float*)d_in[11];
  const float* qW      = (const float*)d_in[12];
  const float* qb      = (const float*)d_in[13];
  const float* vW      = (const float*)d_in[14];
  const float* vb      = (const float*)d_in[15];
  const float* aW      = (const float*)d_in[16];
  const float* ab      = (const float*)d_in[17];
  const float* a_rel   = (const float*)d_in[18];
  const float* m_rel   = (const float*)d_in[19];
  const float* p_rel   = (const float*)d_in[20];
  const float* skip    = (const float*)d_in[21];
  float* out = (float*)d_out;

  const size_t NDF = (size_t)NNODE * DM;
  const size_t WSZ = (size_t)DM * DM;
  const size_t RSZ = (size_t)NH * DH * DH;

  // byte-carved workspace, 256B-aligned regions
  char* base = (char*)d_ws;
  size_t off = 0;
  auto carve = [&](size_t bytes) -> char* {
    off = (off + 255) & ~(size_t)255;
    char* p = base + off;
    off += bytes;
    return p;
  };
  float*  xA     = (float*)carve(NDF * 4);
  float*  xB     = (float*)carve(NDF * 4);
  float*  qproj  = (float*)carve(NDF * 4);
  float*  edgeT  = (float*)carve((size_t)NE * DM * 4);
  float*  escore = (float*)carve((size_t)NE * NH * 4);
  float*  emax   = (float*)carve((size_t)NNODE * NH * 4);
  float*  eden   = (float*)carve((size_t)NNODE * NH * 4);
  float*  aggb   = (float*)carve(NDF * 4);
  __bf16* pooled = (__bf16*)carve(NDF * 2);
  __bf16* x_bf   = (__bf16*)carve(NDF * 2);
  __bf16* kbf    = (__bf16*)carve(NDF * 2);
  __bf16* vbf    = (__bf16*)carve(NDF * 2);
  __bf16* gtmp   = (__bf16*)carve(NDF * 2);
  __bf16* linWt  = (__bf16*)carve(2 * WSZ * 2);
  __bf16* kWt    = (__bf16*)carve(4 * WSZ * 2);
  __bf16* qWt    = (__bf16*)carve(4 * WSZ * 2);
  __bf16* vWt    = (__bf16*)carve(4 * WSZ * 2);
  __bf16* aWt    = (__bf16*)carve(4 * WSZ * 2);
  __bf16* aRelT  = (__bf16*)carve(4 * RSZ * 2);
  __bf16* mRelT  = (__bf16*)carve(4 * RSZ * 2);

  // 0) weight convert + transpose (tiny: ~3M elements total)
  {
    dim3 g((int)((WSZ + 255) / 256), 2);
    transpose_cvt_kernel<<<g, 256, 0, stream>>>(lin_W, linWt, DM, DM);
    dim3 g4((int)((WSZ + 255) / 256), 4);
    transpose_cvt_kernel<<<g4, 256, 0, stream>>>(kW, kWt, DM, DM);
    transpose_cvt_kernel<<<g4, 256, 0, stream>>>(qW, qWt, DM, DM);
    transpose_cvt_kernel<<<g4, 256, 0, stream>>>(vW, vWt, DM, DM);
    transpose_cvt_kernel<<<g4, 256, 0, stream>>>(aW, aWt, DM, DM);
    dim3 gr(16, 24);  // 24 heads of 64x64 per tensor
    transpose_cvt_kernel<<<gr, 256, 0, stream>>>(a_rel, aRelT, DH, DH);
    transpose_cvt_kernel<<<gr, 256, 0, stream>>>(m_rel, mRelT, DH, DH);
  }

  // 1) pool + normalize (bf16 output feeds GEMM directly)
  pool_norm_kernel<<<NMSG, 128, 0, stream>>>(tok_msg, ids_msg, pooled);
  pool_norm_kernel<<<NCON, 128, 0, stream>>>(tok_con, ids_con,
                                             pooled + (size_t)NMSG * DM);

  // 2) encoder linear + relu -> xA (f32 residual) and x_bf (GEMM input)
  launch_gemm(stream, pooled, DM, nullptr, 0, linWt, DM, lin_b, nullptr, 0,
              nullptr, xA, x_bf, DM, NMSG, DM, DM, 1, 1, 0, 0, 0);
  launch_gemm(stream, pooled + (size_t)NMSG * DM, DM, nullptr, 0, linWt + WSZ,
              DM, lin_b + DM, nullptr, 0, nullptr, xA + (size_t)NMSG * DM,
              x_bf + (size_t)NMSG * DM, DM, NCON, DM, DM, 1, 1, 0, 0, 0);

  // 3) two HGT layers
  float* xcur = xA;
  float* xnext = xB;
  const int egrid = (NE * NH + 255) / 256;
  for (int l = 0; l < 2; ++l) {
    const bool last = (l == 1);
    // projections. last layer needs only edge type 0 (con->msg):
    //   k[con], v[con], q[msg]; everything else is dead work.
    launch_gemm(stream, x_bf, DM, nullptr, 0, qWt + (size_t)(l * 2 + 0) * WSZ,
                DM, qb + (size_t)(l * 2 + 0) * DM, nullptr, 0, nullptr, qproj,
                nullptr, DM, NMSG, DM, DM, 0, 1, 0, 0, 0);
    launch_gemm(stream, x_bf + (size_t)NMSG * DM, DM, nullptr, 0,
                kWt + (size_t)(l * 2 + 1) * WSZ, DM,
                kb + (size_t)(l * 2 + 1) * DM, nullptr, 0, nullptr, nullptr,
                kbf + (size_t)NMSG * DM, DM, NCON, DM, DM, 0, 1, 0, 0, 0);
    launch_gemm(stream, x_bf + (size_t)NMSG * DM, DM, nullptr, 0,
                vWt + (size_t)(l * 2 + 1) * WSZ, DM,
                vb + (size_t)(l * 2 + 1) * DM, nullptr, 0, nullptr, nullptr,
                vbf + (size_t)NMSG * DM, DM, NCON, DM, DM, 0, 1, 0, 0, 0);
    if (!last) {
      launch_gemm(stream, x_bf, DM, nullptr, 0,
                  kWt + (size_t)(l * 2 + 0) * WSZ, DM,
                  kb + (size_t)(l * 2 + 0) * DM, nullptr, 0, nullptr, nullptr,
                  kbf, DM, NMSG, DM, DM, 0, 1, 0, 0, 0);
      launch_gemm(stream, x_bf, DM, nullptr, 0,
                  vWt + (size_t)(l * 2 + 0) * WSZ, DM,
                  vb + (size_t)(l * 2 + 0) * DM, nullptr, 0, nullptr, nullptr,
                  vbf, DM, NMSG, DM, DM, 0, 1, 0, 0, 0);
      launch_gemm(stream, x_bf + (size_t)NMSG * DM, DM, nullptr, 0,
                  qWt + (size_t)(l * 2 + 1) * WSZ, DM,
                  qb + (size_t)(l * 2 + 1) * DM, nullptr, 0, nullptr,
                  qproj + (size_t)NMSG * DM, nullptr, DM, NCON, DM, DM, 0, 1,
                  0, 0, 0);
    }

    const int nE = last ? 1 : 2;
    for (int e = 0; e < nE; ++e) {
      const int* srcI = (e == 0) ? src_cm : src_mc;
      const int* dstI = (e == 0) ? dst_cm : dst_mc;
      const int srcBase = (e == 0) ? NMSG : 0;
      const int dstBase = (e == 0) ? 0 : NMSG;
      const int Nd = (e == 0) ? NMSG : NCON;

      launch_fill(stream, emax, -3.402823466e38f, (long long)NNODE * NH);
      launch_fill(stream, eden, 0.f, (long long)NNODE * NH);

      // ke = gather(k, src) @ a_rel[l][e]   (per-head 64x64, gather fused)
      launch_gemm(stream, kbf, DM, srcI, srcBase,
                  aRelT + (size_t)(l * 2 + e) * RSZ, DH, nullptr, nullptr, 0,
                  nullptr, edgeT, nullptr, DM, NE, DH, DH, 0, NH, DH, DH * DH,
                  DH);
      score_max_kernel<<<egrid, 256, 0, stream>>>(
          qproj, dstBase, dstI, edgeT, p_rel + (size_t)(l * 2 + e) * NH,
          escore, emax, NE);
      exp_den_kernel<<<egrid, 256, 0, stream>>>(dstI, dstBase, escore, emax,
                                                eden, NE);
      // msg = gather(v, src) @ m_rel[l][e]  (reuse edgeT)
      launch_gemm(stream, vbf, DM, srcI, srcBase,
                  mRelT + (size_t)(l * 2 + e) * RSZ, DH, nullptr, nullptr, 0,
                  nullptr, edgeT, nullptr, DM, NE, DH, DH, 0, NH, DH, DH * DH,
                  DH);
      launch_fill(stream, aggb + (size_t)dstBase * DM, 0.f, (long long)Nd * DM);
      aggregate_kernel<<<65535, 256, 0, stream>>>(dstI, dstBase, edgeT, escore,
                                                  eden, aggb,
                                                  (long long)NE * DM);
    }

    // output update: o = gelu(agg) @ aW + ab; x' = sig(skip)*o + (1-sig)*x
    const int nT = last ? 1 : 2;
    for (int t = 0; t < nT; ++t) {
      const int base = (t == 0) ? 0 : NMSG;
      const int Nt = (t == 0) ? NMSG : NCON;
      {
        long long n = (long long)Nt * DM;
        int blocks = (int)((n + 255) / 256);
        if (blocks > 65535) blocks = 65535;
        gelu_bf16_kernel<<<blocks, 256, 0, stream>>>(aggb + (size_t)base * DM,
                                                     gtmp + (size_t)base * DM,
                                                     n);
      }
      float*  Yf = (last && t == 0) ? out : (xnext + (size_t)base * DM);
      __bf16* Yb = last ? nullptr : (x_bf + (size_t)base * DM);
      launch_gemm(stream, gtmp + (size_t)base * DM, DM, nullptr, 0,
                  aWt + (size_t)(l * 2 + t) * WSZ, DM,
                  ab + (size_t)(l * 2 + t) * DM, xcur + (size_t)base * DM, DM,
                  skip + (size_t)(l * 2 + t), Yf, Yb, DM, Nt, DM, DM, 2, 1, 0,
                  0, 0);
    }
    float* tmp = xcur; xcur = xnext; xnext = tmp;
  }
}
